// LoRAAttentionRouter_3135326126143
// MI455X (gfx1250) — matrix operand order repaired and checked by reading, not compiled, not run on hardware
//
#include <hip/hip_runtime.h>

typedef __attribute__((ext_vector_type(16))) __bf16 v16bf;
typedef __attribute__((ext_vector_type(8)))  float  v8f;
typedef __attribute__((ext_vector_type(4)))  float  f32x4;   // native vector (NT-store capable)

#define NB       65536
#define IN_F     1024
#define NC       64
#define WAVES    8
#define ROWS_W   16
#define ROWS_BLK (WAVES * ROWS_W)   // 128 rows per block

// ---------------------------------------------------------------------------
// Prep: attn_w [64][1024] f32  ->  wt [64][1024] bf16 (same layout, in d_ws).
// Each WMMA B-fragment is one contiguous 32B run (16 k) of a wt row.
// ---------------------------------------------------------------------------
__global__ void prep_wt_kernel(const float* __restrict__ w, __bf16* __restrict__ wt) {
    int idx = blockIdx.x * 256 + threadIdx.x;      // 0 .. 65535
    int c = idx >> 10;                             // class
    int k = idx & (IN_F - 1);                      // feature index
    wt[c * IN_F + k] = (__bf16)w[idx];
}

// ---------------------------------------------------------------------------
// Fused router: bf16 WMMA scores -> top2 -> softmax -> weighted lora + residual
// ---------------------------------------------------------------------------
__global__ __launch_bounds__(256) void lora_router_kernel(
    const float*  __restrict__ feat,    // [NB][IN_F] f32
    const __bf16* __restrict__ wt,      // [NC][IN_F] bf16 (converted weights)
    const float*  __restrict__ bias,    // [NC]
    const float*  __restrict__ lora,    // [NC][IN_F] f32
    float*        __restrict__ out)     // [NB][IN_F] f32
{
    __shared__ float s_scores[WAVES][ROWS_W][NC];   // 32 KB
    __shared__ float s_w[WAVES][ROWS_W][2];
    __shared__ int   s_i[WAVES][ROWS_W][2];

    const int lane = threadIdx.x & 31;
    const int wave = threadIdx.x >> 5;
    const int m    = lane & 15;        // A: row-in-tile / B: col N / C: col N
    const int h    = lane >> 4;        // lane-half
    const int row0 = blockIdx.x * ROWS_BLK + wave * ROWS_W;

    // ---- Phase 1: 16x64 score tile via WMMA, K-loop over 1024 -------------
    // Software-pipelined: loads for tile i+1 are issued before the WMMAs of
    // tile i, so L2 latency overlaps XDL work instead of stalling each WMMA.
    v8f acc0 = {}, acc1 = {}, acc2 = {}, acc3 = {};
    const float*  arow  = feat + (size_t)(row0 + m) * IN_F + h * 16;  // A base
    const __bf16* brow  = wt + (size_t)m * IN_F + h * 16;             // B base

    // prologue: tile 0
    f32x4 cf0 = *(const f32x4*)(arow);
    f32x4 cf1 = *(const f32x4*)(arow + 4);
    f32x4 cf2 = *(const f32x4*)(arow + 8);
    f32x4 cf3 = *(const f32x4*)(arow + 12);
    v16bf cb0 = *(const v16bf*)(brow);
    v16bf cb1 = *(const v16bf*)(brow + 16 * IN_F);
    v16bf cb2 = *(const v16bf*)(brow + 32 * IN_F);
    v16bf cb3 = *(const v16bf*)(brow + 48 * IN_F);

    for (int k0 = 0; k0 < IN_F; k0 += 32) {
        // prefetch next tile (wraps to tile 0 on the last iteration: branchless)
        const int kn = (k0 + 32) & (IN_F - 1);
        const float*  an = arow + kn;
        const __bf16* bn = brow + kn;
        f32x4 nf0 = *(const f32x4*)(an);
        f32x4 nf1 = *(const f32x4*)(an + 4);
        f32x4 nf2 = *(const f32x4*)(an + 8);
        f32x4 nf3 = *(const f32x4*)(an + 12);
        v16bf nb0 = *(const v16bf*)(bn);
        v16bf nb1 = *(const v16bf*)(bn + 16 * IN_F);
        v16bf nb2 = *(const v16bf*)(bn + 32 * IN_F);
        v16bf nb3 = *(const v16bf*)(bn + 48 * IN_F);

        // A fragment (contiguous k, same as B): half h=0 -> K=k0+0..15
        //                                     half h=1 -> K=k0+16..31
        v16bf a;
        a[0]=(__bf16)cf0.x;  a[1]=(__bf16)cf0.y;  a[2]=(__bf16)cf0.z;  a[3]=(__bf16)cf0.w;
        a[4]=(__bf16)cf1.x;  a[5]=(__bf16)cf1.y;  a[6]=(__bf16)cf1.z;  a[7]=(__bf16)cf1.w;
        a[8]=(__bf16)cf2.x;  a[9]=(__bf16)cf2.y;  a[10]=(__bf16)cf2.z; a[11]=(__bf16)cf2.w;
        a[12]=(__bf16)cf3.x; a[13]=(__bf16)cf3.y; a[14]=(__bf16)cf3.z; a[15]=(__bf16)cf3.w;

        acc0 = __builtin_amdgcn_wmma_f32_16x16x32_bf16(false, a, false, cb0, (short)0, acc0, false, false);
        acc1 = __builtin_amdgcn_wmma_f32_16x16x32_bf16(false, a, false, cb1, (short)0, acc1, false, false);
        acc2 = __builtin_amdgcn_wmma_f32_16x16x32_bf16(false, a, false, cb2, (short)0, acc2, false, false);
        acc3 = __builtin_amdgcn_wmma_f32_16x16x32_bf16(false, a, false, cb3, (short)0, acc3, false, false);

        cf0 = nf0; cf1 = nf1; cf2 = nf2; cf3 = nf3;
        cb0 = nb0; cb1 = nb1; cb2 = nb2; cb3 = nb3;
    }

    // ---- Bias add + scatter scores to LDS (C layout: N=m, M=v+8h) ---------
    {
        float bv0 = bias[m];
        float bv1 = bias[16 + m];
        float bv2 = bias[32 + m];
        float bv3 = bias[48 + m];
#pragma unroll
        for (int v = 0; v < 8; ++v) {
            int r = v + 8 * h;
            s_scores[wave][r][m]      = acc0[v] + bv0;
            s_scores[wave][r][16 + m] = acc1[v] + bv1;
            s_scores[wave][r][32 + m] = acc2[v] + bv2;
            s_scores[wave][r][48 + m] = acc3[v] + bv3;
        }
    }
    __syncthreads();

    // ---- Phase 2: top-2 + softmax (lanes 0..15 own one row each) ----------
    if (lane < 16) {
        const float* sr = s_scores[wave][lane];
        float v0 = -__builtin_inff(), v1 = -__builtin_inff();
        int i0 = 0, i1 = 0;
#pragma unroll 4
        for (int c = 0; c < NC; ++c) {
            float v = sr[c];
            if (v > v0)      { v1 = v0; i1 = i0; v0 = v; i0 = c; }
            else if (v > v1) { v1 = v;  i1 = c; }
        }
        float e  = __expf(v1 - v0);         // <= 1
        float w1 = e / (1.0f + e);
        float w0 = 1.0f - w1;
        s_w[wave][lane][0] = w0;  s_w[wave][lane][1] = w1;
        s_i[wave][lane][0] = i0;  s_i[wave][lane][1] = i1;
    }
    __syncthreads();

    // ---- Phase 3: out = feat + w0*lora[i0] + w1*lora[i1] ------------------
    // Feature rows were just read by this block -> L2 hit on re-read.
    // Output is write-once: non-temporal stores keep L2 for features/lora.
#pragma unroll 2
    for (int r = 0; r < ROWS_W; ++r) {
        const int row = row0 + r;
        const float w0 = s_w[wave][r][0];
        const float w1 = s_w[wave][r][1];
        const int  i0 = s_i[wave][r][0];
        const int  i1 = s_i[wave][r][1];
        const float* fr = feat + (size_t)row * IN_F;
        const float* l0 = lora + (size_t)i0  * IN_F;
        const float* l1 = lora + (size_t)i1  * IN_F;
        float*       po = out  + (size_t)row * IN_F;
#pragma unroll
        for (int ch = 0; ch < 8; ++ch) {
            const int c = ch * 128 + lane * 4;
            f32x4 f  = *(const f32x4*)(fr + c);
            f32x4 a0 = *(const f32x4*)(l0 + c);
            f32x4 a1 = *(const f32x4*)(l1 + c);
            f32x4 o  = f + w0 * a0 + w1 * a1;
            __builtin_nontemporal_store(o, (f32x4*)(po + c));
        }
    }
}

extern "C" void kernel_launch(void* const* d_in, const int* in_sizes, int n_in,
                              void* d_out, int out_size, void* d_ws, size_t ws_size,
                              hipStream_t stream) {
    (void)in_sizes; (void)n_in; (void)out_size; (void)ws_size;
    const float* feat   = (const float*)d_in[0];   // [65536][1024]
    const float* attn_w = (const float*)d_in[1];   // [64][1024]
    const float* attn_b = (const float*)d_in[2];   // [64]
    const float* lora   = (const float*)d_in[3];   // [64][1024]
    float*       out    = (float*)d_out;
    __bf16*      wt     = (__bf16*)d_ws;           // 128 KB scratch

    prep_wt_kernel<<<(NC * IN_F) / 256, 256, 0, stream>>>(attn_w, wt);
    lora_router_kernel<<<NB / ROWS_BLK, 256, 0, stream>>>(feat, wt, attn_b, lora, out);
}
